// LSTM_model_59399397704120
// MI455X (gfx1250) — matrix operand order repaired and checked by reading, not compiled, not run on hardware
//
#include <hip/hip_runtime.h>

// ---------------------------------------------------------------------------
// Problem constants
// ---------------------------------------------------------------------------
#define BATCH   64
#define SEQ     1024
#define DIN     512
#define HID     1024
#define NGATES  (4 * HID)          // 4096
#define NWG     64                 // persistent workgroups; each owns 16 hidden units
#define TPB     256                // 8 waves of 32
#define BH      (BATCH * HID)      // h-state elements per parity buffer

typedef __bf16 bf16;
typedef __attribute__((ext_vector_type(16))) __bf16 v16bf;
typedef __attribute__((ext_vector_type(8)))  __bf16 v8bf;
typedef __attribute__((ext_vector_type(8)))  float  v8f;

// ---------------------------------------------------------------------------
// Helpers
// ---------------------------------------------------------------------------
__device__ __forceinline__ unsigned short f32_to_bf16(float f) {
  unsigned int u = __float_as_uint(f);
  unsigned int lsb = (u >> 16) & 1u;
  u += 0x7fffu + lsb;                     // round-to-nearest-even
  return (unsigned short)(u >> 16);
}

__device__ __forceinline__ float sigmoidf_(float x) {
  return 1.0f / (1.0f + __expf(-x));
}

// A fragment (16x32 bf16 tile, row-major A with leading dim lda):
// lanes 0-15: row M=lane, K = kb+0..7 and kb+16..23
// lanes 16-31: row M=lane-16, K = kb+8..15 and kb+24..31
__device__ __forceinline__ v16bf load_a_frag(const bf16* A, size_t lda,
                                             int mrow, int kb, int lane) {
  int m  = mrow + (lane & 15);
  int k0 = kb + ((lane >> 4) << 3);       // +8 for upper half-wave
  const bf16* p = A + (size_t)m * lda + (size_t)k0;
  v8bf lo = *reinterpret_cast<const v8bf*>(p);
  v8bf hi = *reinterpret_cast<const v8bf*>(p + 16);
  return __builtin_shufflevector(lo, hi, 0,1,2,3,4,5,6,7,8,9,10,11,12,13,14,15);
}

// B fragment (32x16 bf16, B[k][n] = W[n][k], W row-major with leading dim ldw):
// lanes 0-15: col N=lane, K = kb+0..7 and kb+16..23 ; lanes 16-31: col N=lane-16, K = kb+8..15 and kb+24..31
__device__ __forceinline__ v16bf load_b_frag(const bf16* W, int ldw,
                                             int nbase, int kb, int lane) {
  int n  = nbase + (lane & 15);
  int k0 = kb + ((lane >> 4) << 3);       // +8 for upper half-wave
  const bf16* p = W + (size_t)n * (size_t)ldw + (size_t)k0;
  v8bf lo = *reinterpret_cast<const v8bf*>(p);
  v8bf hi = *reinterpret_cast<const v8bf*>(p + 16);
  return __builtin_shufflevector(lo, hi, 0,1,2,3,4,5,6,7,8,9,10,11,12,13,14,15);
}

// Two M-tiles (mrow0, mrow0+16) x one N-tile, K-loop in steps of 32.
template <int K>
__device__ __forceinline__ void gemm2(v8f& acc0, v8f& acc1,
                                      const bf16* A, size_t lda, int mrow0,
                                      const bf16* W, int ldw, int nbase, int lane) {
#pragma unroll 4
  for (int kb = 0; kb < K; kb += 32) {
    v16bf b  = load_b_frag(W, ldw, nbase, kb, lane);
    v16bf a0 = load_a_frag(A, lda, mrow0,      kb, lane);
    v16bf a1 = load_a_frag(A, lda, mrow0 + 16, kb, lane);
    acc0 = __builtin_amdgcn_wmma_f32_16x16x32_bf16(false, a0, false, b,
                                                   (short)0, acc0, false, false);
    acc1 = __builtin_amdgcn_wmma_f32_16x16x32_bf16(false, a1, false, b,
                                                   (short)0, acc1, false, false);
  }
}

// Dump a 16x16 f32 accumulator tile into shared [64][16] layout.
// VGPR r: lanes 0-15 hold (M=r, N=lane), lanes 16-31 hold (M=8+r, N=lane-16).
__device__ __forceinline__ void store_tile(float* base16, const v8f& c,
                                           int mt, int lane) {
  int j  = lane & 15;
  int mh = (lane >> 4) * 8;
#pragma unroll
  for (int r = 0; r < 8; ++r)
    base16[(mt * 16 + mh + r) * 16 + j] = c[r];
}

__device__ __forceinline__ void grid_barrier(int* cnt, int target) {
  __syncthreads();
  __threadfence();
  if (threadIdx.x == 0) {
    __hip_atomic_fetch_add(cnt, 1, __ATOMIC_RELEASE, __HIP_MEMORY_SCOPE_AGENT);
    while (__hip_atomic_load(cnt, __ATOMIC_ACQUIRE, __HIP_MEMORY_SCOPE_AGENT) < target) {
      __builtin_amdgcn_s_sleep(2);
    }
  }
  __syncthreads();
}

// ---------------------------------------------------------------------------
// Utility kernels
// ---------------------------------------------------------------------------
__global__ void cvt_f32_bf16_kernel(const float* __restrict__ in,
                                    unsigned short* __restrict__ out, int n) {
  int i = blockIdx.x * blockDim.x + threadIdx.x;
  if (i < n) out[i] = f32_to_bf16(in[i]);
}

__global__ void init_state_kernel(int* cnt, unsigned short* h0,
                                  unsigned short* h1, int nh) {
  int i = blockIdx.x * blockDim.x + threadIdx.x;
  if (i == 0) *cnt = 0;
  for (int k = i; k < nh; k += gridDim.x * blockDim.x) {
    h0[k] = 0;
    h1[k] = 0;
  }
}

__global__ void head_reduce_kernel(const float* __restrict__ partials,
                                   const float* __restrict__ bm2,
                                   float* __restrict__ out) {
  int idx = blockIdx.x * blockDim.x + threadIdx.x;   // 65536 = B*L
  int m = idx >> 10;
  int t = idx & 1023;
  float s = bm2[0];
#pragma unroll 8
  for (int g = 0; g < NWG; ++g)
    s += partials[(size_t)t * (NWG * BATCH) + g * BATCH + m];
  out[(size_t)m * SEQ + t] = s;
}

// ---------------------------------------------------------------------------
// Persistent fused 2-layer LSTM + MLP head
// ---------------------------------------------------------------------------
__global__ __launch_bounds__(TPB, 1) void lstm_persistent_kernel(
    const unsigned short* __restrict__ xbf,     // (B, L, DIN) bf16
    const unsigned short* __restrict__ wih0,    // (4H, DIN)  bf16
    const unsigned short* __restrict__ whh0,    // (4H, H)    bf16
    const unsigned short* __restrict__ wih1,    // (4H, H)    bf16
    const unsigned short* __restrict__ whh1,    // (4H, H)    bf16
    const unsigned short* __restrict__ w1,      // (H, H)     bf16
    const float* __restrict__ bih0, const float* __restrict__ bhh0,
    const float* __restrict__ bih1, const float* __restrict__ bhh1,
    const float* __restrict__ bm1,  const float* __restrict__ w2,
    unsigned short* __restrict__ h0,            // ping-pong (2, B, H) bf16
    unsigned short* __restrict__ h1,            // ping-pong (2, B, H) bf16
    float* __restrict__ partials,               // (L, NWG, B)
    int* __restrict__ cnt) {
  const int tid  = threadIdx.x;
  const int wid  = tid >> 5;
  const int lane = tid & 31;
  const int g    = blockIdx.x;      // hidden slice owner
  const int j0   = g * 16;          // hidden columns [j0, j0+16)
  const int q    = wid >> 1;        // gate index 0..3 (i,f,g,o)
  const int mr0  = (wid & 1) * 32;  // first of two M-tile rows (0 or 32)

  const bf16* xb   = reinterpret_cast<const bf16*>(xbf);
  const bf16* Wih0 = reinterpret_cast<const bf16*>(wih0);
  const bf16* Whh0 = reinterpret_cast<const bf16*>(whh0);
  const bf16* Wih1 = reinterpret_cast<const bf16*>(wih1);
  const bf16* Whh1 = reinterpret_cast<const bf16*>(whh1);
  const bf16* W1b  = reinterpret_cast<const bf16*>(w1);
  const bf16* h0b  = reinterpret_cast<const bf16*>(h0);
  const bf16* h1b  = reinterpret_cast<const bf16*>(h1);

  __shared__ float ldsg[4 * 64 * 16];   // [gate][m][j] staging (16 KB)

  float c0[4] = {0.f, 0.f, 0.f, 0.f};
  float c1[4] = {0.f, 0.f, 0.f, 0.f};
  int bar = 0;

  const int nbase = q * HID + j0;       // gate-block column base

  for (int t = 0; t < SEQ; ++t) {
    const int p  = t & 1;
    const int pn = p ^ 1;

    // ---------------- layer 0: gates = x_t @ Wih0^T + h0_prev @ Whh0^T ----
    {
      v8f acc0 = {}, acc1 = {};
      gemm2<DIN>(acc0, acc1, xb + (size_t)t * DIN, (size_t)SEQ * DIN, mr0,
                 Wih0, DIN, nbase, lane);
      gemm2<HID>(acc0, acc1, h0b + (size_t)p * BH, HID, mr0,
                 Whh0, HID, nbase, lane);
      store_tile(ldsg + q * 1024, acc0, mr0 >> 4, lane);
      store_tile(ldsg + q * 1024, acc1, (mr0 >> 4) + 1, lane);
    }
    __syncthreads();
#pragma unroll
    for (int i = 0; i < 4; ++i) {
      int idx = tid * 4 + i;
      int m = idx >> 4, j = idx & 15, jj = j0 + j;
      float xi = ldsg[0 * 1024 + m * 16 + j] + bih0[jj]           + bhh0[jj];
      float xf = ldsg[1 * 1024 + m * 16 + j] + bih0[HID + jj]     + bhh0[HID + jj];
      float xg = ldsg[2 * 1024 + m * 16 + j] + bih0[2 * HID + jj] + bhh0[2 * HID + jj];
      float xo = ldsg[3 * 1024 + m * 16 + j] + bih0[3 * HID + jj] + bhh0[3 * HID + jj];
      float ig = sigmoidf_(xi), fg = sigmoidf_(xf);
      float gg = tanhf(xg),     og = sigmoidf_(xo);
      c0[i] = fg * c0[i] + ig * gg;
      float hv = og * tanhf(c0[i]);
      h0[(size_t)pn * BH + (size_t)m * HID + jj] = f32_to_bf16(hv);
    }
    bar += NWG;
    grid_barrier(cnt, bar);   // publish h0_t to all workgroups

    // ---------------- layer 1: gates = h0_t @ Wih1^T + h1_prev @ Whh1^T ---
    {
      v8f acc0 = {}, acc1 = {};
      gemm2<HID>(acc0, acc1, h0b + (size_t)pn * BH, HID, mr0,
                 Wih1, HID, nbase, lane);
      gemm2<HID>(acc0, acc1, h1b + (size_t)p * BH, HID, mr0,
                 Whh1, HID, nbase, lane);
      store_tile(ldsg + q * 1024, acc0, mr0 >> 4, lane);
      store_tile(ldsg + q * 1024, acc1, (mr0 >> 4) + 1, lane);
    }
    __syncthreads();
#pragma unroll
    for (int i = 0; i < 4; ++i) {
      int idx = tid * 4 + i;
      int m = idx >> 4, j = idx & 15, jj = j0 + j;
      float xi = ldsg[0 * 1024 + m * 16 + j] + bih1[jj]           + bhh1[jj];
      float xf = ldsg[1 * 1024 + m * 16 + j] + bih1[HID + jj]     + bhh1[HID + jj];
      float xg = ldsg[2 * 1024 + m * 16 + j] + bih1[2 * HID + jj] + bhh1[2 * HID + jj];
      float xo = ldsg[3 * 1024 + m * 16 + j] + bih1[3 * HID + jj] + bhh1[3 * HID + jj];
      float ig = sigmoidf_(xi), fg = sigmoidf_(xf);
      float gg = tanhf(xg),     og = sigmoidf_(xo);
      c1[i] = fg * c1[i] + ig * gg;
      float hv = og * tanhf(c1[i]);
      h1[(size_t)pn * BH + (size_t)m * HID + jj] = f32_to_bf16(hv);
    }
    bar += NWG;
    grid_barrier(cnt, bar);   // publish h1_t to all workgroups

    // ---------------- MLP head slice: y = relu(h1_t @ W1^T + bm1) ---------
    if (wid < 4) {            // 4 M-tiles, one per wave (uniform branch)
      v8f acc = {};
#pragma unroll 4
      for (int kb = 0; kb < HID; kb += 32) {
        v16bf b = load_b_frag(W1b, HID, j0, kb, lane);
        v16bf a = load_a_frag(h1b + (size_t)pn * BH, HID, wid * 16, kb, lane);
        acc = __builtin_amdgcn_wmma_f32_16x16x32_bf16(false, a, false, b,
                                                      (short)0, acc, false, false);
      }
      store_tile(ldsg, acc, wid, lane);
    }
    __syncthreads();
    if (tid < BATCH) {
      float s = 0.f;
#pragma unroll
      for (int j = 0; j < 16; ++j) {
        float y = ldsg[tid * 16 + j] + bm1[j0 + j];
        y = fmaxf(y, 0.f);
        s += y * w2[j0 + j];
      }
      partials[(size_t)t * (NWG * BATCH) + g * BATCH + tid] = s;
    }
    __syncthreads();          // protect ldsg before next step's dump
  }
}

// ---------------------------------------------------------------------------
// Host launcher
// ---------------------------------------------------------------------------
extern "C" void kernel_launch(void* const* d_in, const int* in_sizes, int n_in,
                              void* d_out, int out_size, void* d_ws, size_t ws_size,
                              hipStream_t stream) {
  (void)in_sizes; (void)n_in; (void)out_size; (void)ws_size;
  const float* x    = (const float*)d_in[0];
  const float* Wih0 = (const float*)d_in[1];
  const float* Whh0 = (const float*)d_in[2];
  const float* bih0 = (const float*)d_in[3];
  const float* bhh0 = (const float*)d_in[4];
  const float* Wih1 = (const float*)d_in[5];
  const float* Whh1 = (const float*)d_in[6];
  const float* bih1 = (const float*)d_in[7];
  const float* bhh1 = (const float*)d_in[8];
  const float* W1   = (const float*)d_in[9];
  const float* bm1  = (const float*)d_in[10];
  const float* W2   = (const float*)d_in[11];
  const float* bm2  = (const float*)d_in[12];

  char* ws = (char*)d_ws;
  size_t off = 0;
  auto take = [&](size_t bytes) -> char* {
    char* r = ws + off;
    off = (off + bytes + 255) & ~(size_t)255;
    return r;
  };

  int*            cnt    = (int*)take(sizeof(int));
  unsigned short* h0     = (unsigned short*)take((size_t)2 * BH * 2);
  unsigned short* h1     = (unsigned short*)take((size_t)2 * BH * 2);
  unsigned short* xbf    = (unsigned short*)take((size_t)BATCH * SEQ * DIN * 2);
  unsigned short* wih0b  = (unsigned short*)take((size_t)NGATES * DIN * 2);
  unsigned short* whh0b  = (unsigned short*)take((size_t)NGATES * HID * 2);
  unsigned short* wih1b  = (unsigned short*)take((size_t)NGATES * HID * 2);
  unsigned short* whh1b  = (unsigned short*)take((size_t)NGATES * HID * 2);
  unsigned short* w1b    = (unsigned short*)take((size_t)HID * HID * 2);
  float*          parts  = (float*)take((size_t)SEQ * NWG * BATCH * 4);

  auto cvt = [&](const float* in, unsigned short* out, int n) {
    cvt_f32_bf16_kernel<<<(n + 255) / 256, 256, 0, stream>>>(in, out, n);
  };
  cvt(x,    xbf,   BATCH * SEQ * DIN);
  cvt(Wih0, wih0b, NGATES * DIN);
  cvt(Whh0, whh0b, NGATES * HID);
  cvt(Wih1, wih1b, NGATES * HID);
  cvt(Whh1, whh1b, NGATES * HID);
  cvt(W1,   w1b,   HID * HID);

  init_state_kernel<<<256, 256, 0, stream>>>(cnt, h0, h1, 2 * BH);

  lstm_persistent_kernel<<<NWG, TPB, 0, stream>>>(
      xbf, wih0b, whh0b, wih1b, whh1b, w1b,
      bih0, bhh0, bih1, bhh1, bm1, W2,
      h0, h1, parts, cnt);

  head_reduce_kernel<<<(BATCH * SEQ) / 256, 256, 0, stream>>>(
      parts, bm2, (float*)d_out);
}
